// Network_987842478348
// MI455X (gfx1250) — matrix operand order repaired and checked by reading, not compiled, not run on hardware
//
#include <hip/hip_runtime.h>
#include <hip/hip_bf16.h>
#include <cstddef>

// ---------------------------------------------------------------------------
// MI455X (gfx1250) implementation.
// Memory-bound phase: two streaming passes over hidden_state (2x201MB ~ 17us
// at 23.3 TB/s). Compute phase: all dense GEMMs ([4224,768]x[768,200] and four
// [4224,200]x[200,{200,600}]) run through v_wmma_f32_16x16x32_f16 (f16 inputs,
// f32 accumulate), one 16x16 tile per wave32, operands loaded per the CDNA5
// WMMA VGPR layouts. Tiny ops (33x33 attention, LayerNorm, classifier) are
// plain VALU.
// ---------------------------------------------------------------------------

typedef _Float16 v16h __attribute__((ext_vector_type(16)));
typedef _Float16 v8h  __attribute__((ext_vector_type(8)));
typedef float    v8f  __attribute__((ext_vector_type(8)));

union AFrag { v16h v; struct { v8h lo, hi; } p; };

#define B_  128
#define L_  512
#define D_  768
#define S_  33
#define DM_ 200
#define DMP 224   // DM padded to x32 (K of small GEMMs) and x16 (N pad)
#define MROWS (B_ * S_)   // 4224 = 264 * 16

// ---------------------------------------------------------------------------
// Weight f32[N,K] -> f16[Npad,Kp], zero padded.
// ---------------------------------------------------------------------------
__global__ void wconv_kernel(const float* __restrict__ in, _Float16* __restrict__ out,
                             int N, int K, int Kp, int total) {
    int idx = blockIdx.x * 256 + threadIdx.x;
    if (idx >= total) return;
    int n = idx / Kp, k = idx % Kp;
    float v = (n < N && k < K) ? in[n * K + k] : 0.f;
    out[idx] = (_Float16)v;
}

// ---------------------------------------------------------------------------
// scores[b,l] = dot(h[b,l,:], fc5_w) + fc5_b.  One wave per token.
// ---------------------------------------------------------------------------
__global__ void score_kernel(const float* __restrict__ h, const float* __restrict__ w,
                             const float* __restrict__ bptr, float* __restrict__ scores) {
    __shared__ float wl[D_];
    int tid = threadIdx.x;
    wl[tid] = w[tid]; wl[tid + 256] = w[tid + 256]; wl[tid + 512] = w[tid + 512];
    __syncthreads();
    int wave = tid >> 5, lane = tid & 31;
    size_t t = (size_t)blockIdx.x * 8 + wave;          // 0..65535
    const float* row = h + t * D_;
    float s = 0.f;
    #pragma unroll
    for (int k = 0; k < 24; ++k) {                     // 128B coalesced per wave/iter
        int j = k * 32 + lane;
        s += row[j] * wl[j];
    }
    for (int m = 16; m; m >>= 1) s += __shfl_xor(s, m, 32);
    if (lane == 0) scores[t] = s + bptr[0];
}

// ---------------------------------------------------------------------------
// Per-segment softmax + pooling. Block = (b, 256-channel chunk); thread owns
// one channel and streams l=0..511 with segment-flush (boundaries strictly
// increasing, all segments non-empty). Writes pooled as f16 [4224,768].
// ---------------------------------------------------------------------------
__global__ void pool_kernel(const float* __restrict__ h, const int* __restrict__ cidx,
                            const float* __restrict__ scores, _Float16* __restrict__ pooled) {
    __shared__ int   cb[32];
    __shared__ float sc[L_];
    __shared__ float wt[L_];
    int b = blockIdx.x / 3;
    int dbase = (blockIdx.x % 3) * 256;
    int tid = threadIdx.x;
    if (tid < 32) cb[tid] = cidx[b * 32 + tid];
    sc[tid]       = scores[b * L_ + tid];
    sc[tid + 256] = scores[b * L_ + 256 + tid];
    __syncthreads();
    if (tid < S_) {                                    // segment softmax
        int st = (tid == 0) ? 0 : cb[tid - 1];
        int en = (tid == 32) ? L_ : cb[tid];
        float m = -1e30f;
        for (int l = st; l < en; ++l) m = fmaxf(m, sc[l]);
        float s = 0.f;
        for (int l = st; l < en; ++l) { float e = __expf(sc[l] - m); wt[l] = e; s += e; }
        float inv = 1.f / s;
        for (int l = st; l < en; ++l) wt[l] *= inv;
    }
    __syncthreads();
    int d = dbase + tid;
    const float* hb = h + ((size_t)b * L_) * D_ + d;
    float acc = 0.f; int cur = 0;
    for (int l = 0; l < L_; ++l) {
        while (cur < 32 && l >= cb[cur]) {             // flush finished segment
            pooled[((size_t)(b * S_ + cur)) * D_ + d] = (_Float16)acc;
            acc = 0.f; ++cur;
        }
        acc += wt[l] * hb[(size_t)l * D_];             // 1KB coalesced per l
    }
    pooled[((size_t)(b * S_ + cur)) * D_ + d] = (_Float16)acc;  // cur == 32
}

// ---------------------------------------------------------------------------
// WMMA GEMM: C[m,n] = sum_k A[m,k]*W[n,k] + bias[n].  A f16 [4224,Kp],
// W f16 [Ntiles*16, Kp].  One 16x16 output tile per wave, K-loop step 32.
// Fragment layouts per CDNA5 ISA 7.12.2 (16-bit A 16x32 / B 32x16, wave32).
// ---------------------------------------------------------------------------
__global__ void gemm_wmma_kernel(const _Float16* __restrict__ A, const _Float16* __restrict__ W,
                                 const float* __restrict__ bias, int Nbias,
                                 float* __restrict__ outF, int strideF,
                                 _Float16* __restrict__ outH, int strideH,
                                 int Kp, int Ntiles, int relu) {
    int wid  = (blockIdx.x * 256 + threadIdx.x) >> 5;
    int lane = threadIdx.x & 31;
    int mtile = wid / Ntiles, ntile = wid % Ntiles;
    int hf = lane >> 4, l15 = lane & 15;
    const _Float16* Arow = A + (size_t)(mtile * 16 + l15) * Kp;   // A row m = lane&15
    const _Float16* Wrow = W + (size_t)(ntile * 16 + l15) * Kp;   // B col n = lane&15
    v8f acc = {};
    for (int k0 = 0; k0 < Kp; k0 += 32) {
        AFrag a;
        a.p.lo = *(const v8h*)(Arow + k0 + hf * 8);        // K:  0..7  / 8..15
        a.p.hi = *(const v8h*)(Arow + k0 + 16 + hf * 8);   // K: 16..23 / 24..31
        AFrag b;
        b.p.lo = *(const v8h*)(Wrow + k0 + hf * 8);        // K:  0..7  / 8..15
        b.p.hi = *(const v8h*)(Wrow + k0 + 16 + hf * 8);   // K: 16..23 / 24..31
        v16h bfr = b.v;
        acc = __builtin_amdgcn_wmma_f32_16x16x32_f16(false, a.v, false, bfr,
                                                     (short)0, acc, false, false);
    }
    int n = ntile * 16 + l15;
    float bval = (n < Nbias) ? bias[n] : 0.f;
    #pragma unroll
    for (int j = 0; j < 8; ++j) {                          // row = j + 8*hf
        int m = mtile * 16 + j + hf * 8;
        float v = acc[j] + bval;
        if (relu) v = v > 0.f ? v : 0.f;
        if (outF && n < Nbias) outF[(size_t)m * strideF + n] = v;
        if (outH)              outH[(size_t)m * strideH + n] = (_Float16)v;
    }
}

// ---------------------------------------------------------------------------
// Single-head attention per batch row (33 tokens, d=200). q,k in LDS, v via L2.
// Writes att as f16 [4224,224] (pad zeroed).
// ---------------------------------------------------------------------------
__global__ void attn_kernel(const float* __restrict__ qkv, _Float16* __restrict__ attH) {
    __shared__ float qs[S_ * DM_];
    __shared__ float ks[S_ * DM_];
    __shared__ float lg[S_ * S_];
    int b = blockIdx.x, tid = threadIdx.x;
    for (int p = tid; p < S_ * 400; p += 256) {
        int i = p / 400, c = p % 400;
        float v = qkv[((size_t)(b * S_ + i)) * 600 + c];
        if (c < DM_) qs[i * DM_ + c] = v; else ks[i * DM_ + (c - DM_)] = v;
    }
    __syncthreads();
    for (int p = tid; p < S_ * S_; p += 256) {
        int i = p / S_, j = p % S_;
        float s = 0.f;
        for (int c = 0; c < DM_; ++c) s += qs[i * DM_ + c] * ks[j * DM_ + c];
        lg[p] = s * 0.07071067811865475f;                  // 1/sqrt(200)
    }
    __syncthreads();
    if (tid < S_) {
        float m = -1e30f;
        for (int j = 0; j < S_; ++j) m = fmaxf(m, lg[tid * S_ + j]);
        float s = 0.f;
        for (int j = 0; j < S_; ++j) { float e = __expf(lg[tid * S_ + j] - m); lg[tid * S_ + j] = e; s += e; }
        float inv = 1.f / s;
        for (int j = 0; j < S_; ++j) lg[tid * S_ + j] *= inv;
    }
    __syncthreads();
    for (int p = tid; p < S_ * DMP; p += 256) {
        int i = p / DMP, d = p % DMP;
        float v = 0.f;
        if (d < DM_)
            for (int j = 0; j < S_; ++j)
                v += lg[i * S_ + j] * qkv[((size_t)(b * S_ + j)) * 600 + 400 + d];
        attH[((size_t)(b * S_ + i)) * DMP + d] = (_Float16)v;
    }
}

// ---------------------------------------------------------------------------
// Fused residual + LayerNorm over 200. One wave per row; emits f32 [.,200]
// and optional f16 [.,224] (pad zeroed).
// ---------------------------------------------------------------------------
__global__ void ln_kernel(const float* __restrict__ a, const float* __restrict__ res,
                          const float* __restrict__ g, const float* __restrict__ beta,
                          float* __restrict__ outF, _Float16* __restrict__ outH) {
    int row  = (blockIdx.x * 256 + threadIdx.x) >> 5;
    int lane = threadIdx.x & 31;
    const float* ra = a + (size_t)row * DM_;
    const float* rb = res + (size_t)row * DM_;
    float x[7], sum = 0.f;
    #pragma unroll
    for (int i = 0; i < 7; ++i) {
        int j = i * 32 + lane;
        float v = 0.f;
        if (j < DM_) v = ra[j] + rb[j];
        x[i] = v; sum += v;
    }
    for (int m = 16; m; m >>= 1) sum += __shfl_xor(sum, m, 32);
    float mean = sum * (1.f / DM_);
    float vs = 0.f;
    #pragma unroll
    for (int i = 0; i < 7; ++i) { int j = i * 32 + lane; if (j < DM_) { float d = x[i] - mean; vs += d * d; } }
    for (int m = 16; m; m >>= 1) vs += __shfl_xor(vs, m, 32);
    float rinv = rsqrtf(vs * (1.f / DM_) + 1e-5f);
    #pragma unroll
    for (int i = 0; i < 7; ++i) {
        int j = i * 32 + lane;
        if (j < DM_) {
            float v = (x[i] - mean) * rinv * g[j] + beta[j];
            outF[(size_t)row * DM_ + j] = v;
            if (outH) outH[(size_t)row * DMP + j] = (_Float16)v;
        } else if (outH) {                                  // j in [200,224)
            outH[(size_t)row * DMP + j] = (_Float16)0.f;
        }
    }
}

// ---------------------------------------------------------------------------
// Classifier: out[b,s-1,c] = fc_w[c,:200].x[b,0] + fc_w[c,200:].x[b,s] + fc_b[c]
// ---------------------------------------------------------------------------
__global__ void cls_kernel(const float* __restrict__ x2, const float* __restrict__ fw,
                           const float* __restrict__ fb, float* __restrict__ out) {
    int b = blockIdx.x, tid = threadIdx.x;                 // 64 threads
    int s1 = tid >> 1, c = tid & 1;
    const float* r0 = x2 + (size_t)(b * S_) * DM_;
    const float* rs = x2 + (size_t)(b * S_ + s1 + 1) * DM_;
    float a = 0.f;
    for (int j = 0; j < DM_; ++j)
        a += fw[c * 400 + j] * r0[j] + fw[c * 400 + 200 + j] * rs[j];
    out[((size_t)(b * 32 + s1)) * 2 + c] = a + fb[c];
}

// ---------------------------------------------------------------------------
extern "C" void kernel_launch(void* const* d_in, const int* in_sizes, int n_in,
                              void* d_out, int out_size, void* d_ws, size_t ws_size,
                              hipStream_t stream) {
    const float* h        = (const float*)d_in[0];
    const int*   cidx     = (const int*)  d_in[1];
    const float* fc5_w    = (const float*)d_in[2];
    const float* fc5_b    = (const float*)d_in[3];
    const float* fc1_w    = (const float*)d_in[4];
    const float* fc1_b    = (const float*)d_in[5];
    const float* ain_w    = (const float*)d_in[6];
    const float* ain_b    = (const float*)d_in[7];
    const float* aout_w   = (const float*)d_in[8];
    const float* aout_b   = (const float*)d_in[9];
    const float* ln1_g    = (const float*)d_in[10];
    const float* ln1_b    = (const float*)d_in[11];
    const float* lin1_w   = (const float*)d_in[12];
    const float* lin1_b   = (const float*)d_in[13];
    const float* lin2_w   = (const float*)d_in[14];
    const float* lin2_b   = (const float*)d_in[15];
    const float* ln2_g    = (const float*)d_in[16];
    const float* ln2_b    = (const float*)d_in[17];
    const float* fc_w     = (const float*)d_in[18];
    const float* fc_b     = (const float*)d_in[19];
    float* out = (float*)d_out;

    char* ws = (char*)d_ws;
    size_t off = 0;
    auto alloc = [&](size_t bytes) { size_t r = off; off += (bytes + 255) & ~(size_t)255; return r; };
    float*    scores   = (float*)   (ws + alloc((size_t)B_ * L_ * 4));
    _Float16* pooledH  = (_Float16*)(ws + alloc((size_t)MROWS * D_ * 2));
    _Float16* fc1w16   = (_Float16*)(ws + alloc((size_t)224 * D_ * 2));
    _Float16* ainw16   = (_Float16*)(ws + alloc((size_t)608 * DMP * 2));
    _Float16* aoutw16  = (_Float16*)(ws + alloc((size_t)208 * DMP * 2));
    _Float16* lin1w16  = (_Float16*)(ws + alloc((size_t)224 * DMP * 2));
    _Float16* lin2w16  = (_Float16*)(ws + alloc((size_t)208 * DMP * 2));
    float*    x1F      = (float*)   (ws + alloc((size_t)MROWS * DM_ * 4));
    _Float16* x1H      = (_Float16*)(ws + alloc((size_t)MROWS * DMP * 2));
    float*    qkvF     = (float*)   (ws + alloc((size_t)MROWS * 600 * 4));
    _Float16* attH     = (_Float16*)(ws + alloc((size_t)MROWS * DMP * 2));
    float*    att2F    = (float*)   (ws + alloc((size_t)MROWS * DM_ * 4));
    float*    xln1F    = (float*)   (ws + alloc((size_t)MROWS * DM_ * 4));
    _Float16* xln1H    = (_Float16*)(ws + alloc((size_t)MROWS * DMP * 2));
    _Float16* ff1H     = (_Float16*)(ws + alloc((size_t)MROWS * DMP * 2));
    float*    ff2F     = (float*)   (ws + alloc((size_t)MROWS * DM_ * 4));
    float*    x2F      = (float*)   (ws + alloc((size_t)MROWS * DM_ * 4));
    (void)ws_size; (void)n_in; (void)in_sizes; (void)out_size;

    // --- weight conversions to padded f16 ---
    auto wc = [&](const float* in, _Float16* o, int N, int K, int Npad, int Kp) {
        int total = Npad * Kp;
        wconv_kernel<<<(total + 255) / 256, 256, 0, stream>>>(in, o, N, K, Kp, total);
    };
    wc(fc1_w,  fc1w16,  200, D_,  224, D_);
    wc(ain_w,  ainw16,  600, DM_, 608, DMP);
    wc(aout_w, aoutw16, 200, DM_, 208, DMP);
    wc(lin1_w, lin1w16, 200, DM_, 224, DMP);
    wc(lin2_w, lin2w16, 200, DM_, 208, DMP);

    // --- streaming phase: scores then segment-softmax pooling ---
    score_kernel<<<(B_ * L_) / 8, 256, 0, stream>>>(h, fc5_w, fc5_b, scores);
    pool_kernel<<<B_ * 3, 256, 0, stream>>>(h, cidx, scores, pooledH);

    // --- WMMA GEMM pipeline ---
    auto gemm = [&](const _Float16* A, const _Float16* W, const float* bias, int Nbias,
                    float* oF, int sF, _Float16* oH, int sH, int Kp, int Ntiles, int relu) {
        int blocks = (MROWS / 16) * Ntiles / 8;            // 264*Ntiles waves, 8 waves/block
        gemm_wmma_kernel<<<blocks, 256, 0, stream>>>(A, W, bias, Nbias, oF, sF, oH, sH, Kp, Ntiles, relu);
    };
    // x1 = pooled @ fc1_w.T + b        [4224,768] x [768,200]
    gemm(pooledH, fc1w16, fc1_b, 200, x1F, DM_, x1H, DMP, D_, 14, 0);
    // qkv = x1 @ attn_in_w.T + b       [4224,224] x [224,600]
    gemm(x1H, ainw16, ain_b, 600, qkvF, 600, nullptr, 0, DMP, 38, 0);
    // attention (per-b, VALU)
    attn_kernel<<<B_, 256, 0, stream>>>(qkvF, attH);
    // att2 = att @ attn_out_w.T + b
    gemm(attH, aoutw16, aout_b, 200, att2F, DM_, nullptr, 0, DMP, 13, 0);
    // xln1 = LN(x1 + att2)
    ln_kernel<<<MROWS / 8, 256, 0, stream>>>(x1F, att2F, ln1_g, ln1_b, xln1F, xln1H);
    // ff1 = relu(xln1 @ lin1_w.T + b)  (f16 only)
    gemm(xln1H, lin1w16, lin1_b, 200, nullptr, 0, ff1H, DMP, DMP, 14, 1);
    // ff2 = ff1 @ lin2_w.T + b
    gemm(ff1H, lin2w16, lin2_b, 200, ff2F, DM_, nullptr, 0, DMP, 13, 0);
    // x2 = LN(xln1 + ff2)
    ln_kernel<<<MROWS / 8, 256, 0, stream>>>(xln1F, ff2F, ln2_g, ln2_b, x2F, nullptr);
    // classifier
    cls_kernel<<<B_, 64, 0, stream>>>(x2F, fc_w, fc_b, out);
}
